// LocalBranch_46437186404433
// MI455X (gfx1250) — matrix operand order repaired and checked by reading, not compiled, not run on hardware
//
#include <hip/hip_runtime.h>
#include <hip/hip_bf16.h>
#include <math.h>

// ---------------------------------------------------------------------------
// Problem constants (reference: B=1024 graphs, 128 nodes, IN=64, HID=256, OUT=128)
#define NNODES   131072      // B*NODES
#define HIDDIM   256
#define INDIM    64
#define OUTDIM   128
#define NGRAPH   1024
#define TAUF     0.25f
#define VOFF     (NGRAPH * OUTDIM)   // offset of w in d_out (v_loc first)

typedef __attribute__((ext_vector_type(16))) __bf16 v16bf;
typedef __attribute__((ext_vector_type(8)))  float  v8f;

union FragAB { v16bf v; unsigned int u[8]; };

__device__ __forceinline__ unsigned short f2bf(float f) {
  unsigned int x = __float_as_uint(f);
  x += 0x7FFFu + ((x >> 16) & 1u);      // round-to-nearest-even
  return (unsigned short)(x >> 16);
}

__device__ __forceinline__ unsigned int packbf(float a, float b) {
  return (unsigned int)f2bf(a) | ((unsigned int)f2bf(b) << 16);
}

// Sum across the 16 lanes of a half-wave (wave32; masks 1,2,4,8 stay in-half)
__device__ __forceinline__ float redrow16(float v) {
  v += __shfl_xor(v, 1, 32);
  v += __shfl_xor(v, 2, 32);
  v += __shfl_xor(v, 4, 32);
  v += __shfl_xor(v, 8, 32);
  return v;
}

// ---------------------------------------------------------------------------
// Convert f32 weights W[K][N] -> bf16 transposed WT[N][K] (B-fragment friendly:
// consecutive-K pairs are one aligned dword).
__global__ void wconv_kernel(const float* __restrict__ src,
                             unsigned short* __restrict__ dst, int K, int N) {
  int idx = blockIdx.x * blockDim.x + threadIdx.x;
  if (idx >= K * N) return;
  int n = idx / K, k = idx % K;
  dst[idx] = f2bf(src[k * N + n]);
}

// ---------------------------------------------------------------------------
// One MLP layer (out dim = 256):
//   X' = [residual? X + ] relu(LN(A @ W + b)),  XB' = bf16(X')
// A operand: bf16 dwords (ABF) when available (blocks), else f32 (stem, lda=64).
// block = 128 threads (4 waves); each block owns 64 rows; each wave owns one
// 16-row M-tile and all 16 N-tiles.
__global__ __launch_bounds__(128) void mlp_layer_kernel(
    const float* __restrict__ A, const unsigned int* __restrict__ ABF,
    int lda, int K,
    const unsigned int* __restrict__ WT,       // bf16 pairs, [n][K/2] dwords
    const float* __restrict__ bias, const float* __restrict__ g,
    const float* __restrict__ beta,
    float* __restrict__ X, unsigned short* __restrict__ XB, int residual)
{
  __shared__ unsigned int sW[4096];            // 16 KB: [col(256)][16] dwords (k-chunk of 32)
  __shared__ float sBias[HIDDIM], sG[HIDDIM], sBeta[HIDDIM];

  const int tid  = threadIdx.x;
  const int wave = tid >> 5;
  const int lane = tid & 31;
  const int m    = lane & 15;                  // A row / B col within tile
  const int half = lane >> 4;                  // K-half selector
  const int rowbase = blockIdx.x * 64 + wave * 16;

  for (int i = tid; i < HIDDIM; i += 128) {
    sBias[i] = bias[i]; sG[i] = g[i]; sBeta[i] = beta[i];
  }

  v8f acc[16];
  for (int nt = 0; nt < 16; ++nt)
    for (int i = 0; i < 8; ++i) acc[nt][i] = 0.0f;

  const int ksteps = K >> 5;
  const int wtK2   = K >> 1;                   // dwords per weight column
  const float2* arow2 = (const float2*)(A + (size_t)(rowbase + m) * lda);
  const unsigned int* abrow =
      ABF ? (ABF + (size_t)(rowbase + m) * (HIDDIM / 2)) : (const unsigned int*)0;

  for (int ks = 0; ks < ksteps; ++ks) {
    __syncthreads();
    for (int i = tid; i < 4096; i += 128) {    // stage 256x32 bf16 weight chunk
      int n = i >> 4, kk = i & 15;
      sW[i] = WT[n * wtK2 + ks * 16 + kk];
    }
    __syncthreads();

    FragAB af;                                 // A 16x32 bf16 fragment
    if (abrow) {
      // direct bf16: two runs of 4 consecutive dwords -> global_load_b128 pairs
      #pragma unroll
      for (int r = 0; r < 8; ++r) {
        int kl2 = (r < 4) ? (half * 4 + r) : (8 + half * 4 + (r - 4));
        af.u[r] = abrow[ks * 16 + kl2];
      }
    } else {
      // stem: convert f32 H on the fly (K=64 only)
      #pragma unroll
      for (int r = 0; r < 8; ++r) {
        int kl2 = (r < 4) ? (half * 4 + r) : (8 + half * 4 + (r - 4));
        float2 av = arow2[ks * 16 + kl2];
        af.u[r] = packbf(av.x, av.y);
      }
    }

    #pragma unroll
    for (int nt = 0; nt < 16; ++nt) {
      FragAB bfr;                              // B 32x16 fragment from LDS
      int coldw = (nt * 16 + m) * 16 + half * 8;
      #pragma unroll
      for (int r = 0; r < 8; ++r) bfr.u[r] = sW[coldw - half * 4 + ((r < 4) ? r : (4 + r))];
      acc[nt] = __builtin_amdgcn_wmma_f32_16x16x32_bf16(
          false, af.v, false, bfr.v, (short)0, acc[nt], false, false);
    }
  }

  // bias
  #pragma unroll
  for (int nt = 0; nt < 16; ++nt) {
    float b = sBias[nt * 16 + m];
    #pragma unroll
    for (int r = 0; r < 8; ++r) acc[nt][r] += b;
  }
  // per-row LayerNorm + relu (+residual); C layout: row = r + 8*half
  #pragma unroll
  for (int r = 0; r < 8; ++r) {
    float s = 0.f, ss = 0.f;
    #pragma unroll
    for (int nt = 0; nt < 16; ++nt) { float v = acc[nt][r]; s += v; ss += v * v; }
    s = redrow16(s); ss = redrow16(ss);
    float mu   = s * (1.0f / 256.0f);
    float var  = ss * (1.0f / 256.0f) - mu * mu;
    float rstd = rsqrtf(var + 1e-5f);
    int row = rowbase + r + 8 * half;
    float* xrow = X + (size_t)row * HIDDIM;
    unsigned short* xbrow = XB + (size_t)row * HIDDIM;
    #pragma unroll
    for (int nt = 0; nt < 16; ++nt) {
      int col = nt * 16 + m;
      float v = (acc[nt][r] - mu) * rstd * sG[col] + sBeta[col];
      v = fmaxf(v, 0.0f);
      if (residual) v += xrow[col];
      xrow[col]  = v;
      xbrow[col] = f2bf(v);
    }
  }
}

// ---------------------------------------------------------------------------
// Head + per-graph reduction, one workgroup per graph (128 nodes):
// z = LN(xb @ W_out + b); d_ij = ||z_i - z_j||; s_i = mean_j d_ij;
// w = softmax(s/tau); v = sum_i w_i z_i.
#define ZP 132   // z LDS pitch in bf16 (padded against bank conflicts)

__global__ __launch_bounds__(128) void head_reduce_kernel(
    const unsigned int* __restrict__ XB,       // bf16 activations [row][128] dwords
    const unsigned int* __restrict__ WT,       // [n(128)][128] dwords (K=256)
    const float* __restrict__ bias, const float* __restrict__ g,
    const float* __restrict__ beta,
    float* __restrict__ out)
{
  __shared__ unsigned int sW[2048];            // 8 KB weight chunk [col(128)][16]
  __shared__ unsigned short sZ[128 * ZP];      // 33 KB bf16 z tile
  __shared__ float sSq[128], sS[128], sE[128], sWt[128];
  __shared__ float sBias[OUTDIM], sG[OUTDIM], sBeta[OUTDIM];

  const int tid  = threadIdx.x;
  const int wave = tid >> 5;
  const int lane = tid & 31;
  const int m    = lane & 15;
  const int half = lane >> 4;
  const int graph = blockIdx.x;
  const unsigned int* xg = XB + (size_t)graph * 128 * (HIDDIM / 2);

  if (tid < OUTDIM) { sBias[tid] = bias[tid]; sG[tid] = g[tid]; sBeta[tid] = beta[tid]; }

  // ---- head GEMM: each wave owns rows [wave*32, wave*32+32) = 2 M-tiles ----
  v8f acc[2][8];
  for (int mt = 0; mt < 2; ++mt)
    for (int nt = 0; nt < 8; ++nt)
      for (int i = 0; i < 8; ++i) acc[mt][nt][i] = 0.0f;

  for (int ks = 0; ks < 8; ++ks) {
    __syncthreads();
    for (int i = tid; i < 2048; i += 128) {
      int n = i >> 4, kk = i & 15;
      sW[i] = WT[n * 128 + ks * 16 + kk];
    }
    __syncthreads();
    FragAB af[2];
    #pragma unroll
    for (int mt = 0; mt < 2; ++mt) {
      const unsigned int* arow =
          xg + (size_t)(wave * 32 + mt * 16 + m) * (HIDDIM / 2);
      #pragma unroll
      for (int r = 0; r < 8; ++r) {
        int kl2 = (r < 4) ? (half * 4 + r) : (8 + half * 4 + (r - 4));
        af[mt].u[r] = arow[ks * 16 + kl2];
      }
    }
    #pragma unroll
    for (int nt = 0; nt < 8; ++nt) {
      FragAB bfr;
      int coldw = (nt * 16 + m) * 16 + half * 8;
      #pragma unroll
      for (int r = 0; r < 8; ++r) bfr.u[r] = sW[coldw - half * 4 + ((r < 4) ? r : (4 + r))];
      #pragma unroll
      for (int mt = 0; mt < 2; ++mt)
        acc[mt][nt] = __builtin_amdgcn_wmma_f32_16x16x32_bf16(
            false, af[mt].v, false, bfr.v, (short)0, acc[mt][nt], false, false);
    }
  }

  // ---- bias + LN; stash bf16 z and exact f32 row norms in LDS ----
  #pragma unroll
  for (int mt = 0; mt < 2; ++mt) {
    #pragma unroll
    for (int nt = 0; nt < 8; ++nt) {
      float b = sBias[nt * 16 + m];
      #pragma unroll
      for (int r = 0; r < 8; ++r) acc[mt][nt][r] += b;
    }
    #pragma unroll
    for (int r = 0; r < 8; ++r) {
      float s = 0.f, ss = 0.f;
      #pragma unroll
      for (int nt = 0; nt < 8; ++nt) { float v = acc[mt][nt][r]; s += v; ss += v * v; }
      s = redrow16(s); ss = redrow16(ss);
      float mu   = s * (1.0f / 128.0f);
      float var  = ss * (1.0f / 128.0f) - mu * mu;
      float rstd = rsqrtf(var + 1e-5f);
      int lrow = wave * 32 + mt * 16 + r + 8 * half;
      float sq = 0.f;
      #pragma unroll
      for (int nt = 0; nt < 8; ++nt) {
        int col = nt * 16 + m;
        float z = (acc[mt][nt][r] - mu) * rstd * sG[col] + sBeta[col];
        sq += z * z;
        sZ[lrow * ZP + col] = f2bf(z);
      }
      sq = redrow16(sq);
      if (m == 0) sSq[lrow] = sq;
    }
  }
  __syncthreads();

  // ---- Gram via WMMA on LDS z (A rows = z_i, B cols = z_j) + distances ----
  const unsigned int* z32 = (const unsigned int*)sZ;
  #pragma unroll
  for (int mt = 0; mt < 2; ++mt) {
    v8f g2[8];
    for (int nt = 0; nt < 8; ++nt)
      for (int i = 0; i < 8; ++i) g2[nt][i] = 0.0f;
    #pragma unroll
    for (int ks = 0; ks < 4; ++ks) {
      FragAB af;
      int arow = wave * 32 + mt * 16 + m;
      #pragma unroll
      for (int r = 0; r < 8; ++r) {
        int kdw = (r < 4) ? (half * 4 + r) : (8 + half * 4 + (r - 4));
        af.u[r] = z32[arow * (ZP / 2) + ks * 16 + kdw];
      }
      #pragma unroll
      for (int nt = 0; nt < 8; ++nt) {
        FragAB bfr;
        int brow = nt * 16 + m;
        #pragma unroll
        for (int r = 0; r < 8; ++r)
          bfr.u[r] = z32[brow * (ZP / 2) + ks * 16 + ((r < 4) ? (half * 4 + r) : (8 + half * 4 + (r - 4)))];
        g2[nt] = __builtin_amdgcn_wmma_f32_16x16x32_bf16(
            false, af.v, false, bfr.v, (short)0, g2[nt], false, false);
      }
    }
    #pragma unroll
    for (int r = 0; r < 8; ++r) {
      int i = wave * 32 + mt * 16 + r + 8 * half;
      float sqi = sSq[i];
      float dsum = 0.f;
      #pragma unroll
      for (int nt = 0; nt < 8; ++nt) {
        int j = nt * 16 + m;
        float d2 = sqi + sSq[j] - 2.0f * g2[nt][r];
        dsum += sqrtf(fmaxf(d2, 0.0f) + 1e-12f);
      }
      dsum = redrow16(dsum);
      if (m == 0) sS[i] = dsum * (1.0f / 128.0f);
    }
  }
  __syncthreads();

  // ---- softmax over 128 rows (thread t owns row t) ----
  float si = sS[tid];
  float mx = -3.4e38f;
  for (int j = 0; j < 128; ++j) mx = fmaxf(mx, sS[j]);   // LDS broadcast reads
  float e = expf((si - mx) * (1.0f / TAUF));
  sE[tid] = e;
  __syncthreads();
  float Z = 0.f;
  for (int j = 0; j < 128; ++j) Z += sE[j];
  float w = e / Z;
  sWt[tid] = w;
  out[VOFF + graph * OUTDIM + tid] = w;                  // w output
  __syncthreads();

  // ---- v = sum_i w_i z_i ; thread t owns output column t ----
  float v = 0.f;
  for (int row = 0; row < 128; ++row) {
    unsigned int zb = sZ[row * ZP + tid];
    v += sWt[row] * __uint_as_float(zb << 16);
  }
  out[graph * OUTDIM + tid] = v;                         // v_loc output
}

// ---------------------------------------------------------------------------
extern "C" void kernel_launch(void* const* d_in, const int* in_sizes, int n_in,
                              void* d_out, int out_size, void* d_ws, size_t ws_size,
                              hipStream_t stream) {
  (void)in_sizes; (void)n_in; (void)out_size; (void)ws_size;

  const float* H      = (const float*)d_in[0];
  // d_in[1] = batch_ptr (uniform graphs -> unused)
  const float* W_in   = (const float*)d_in[2];
  const float* b_in   = (const float*)d_in[3];
  const float* g_in   = (const float*)d_in[4];
  const float* be_in  = (const float*)d_in[5];
  const float* Wb     = (const float*)d_in[6];
  const float* bb     = (const float*)d_in[7];
  const float* gb     = (const float*)d_in[8];
  const float* betab  = (const float*)d_in[9];
  const float* W_out  = (const float*)d_in[10];
  const float* b_out  = (const float*)d_in[11];
  const float* g_out  = (const float*)d_in[12];
  const float* be_out = (const float*)d_in[13];
  float* out = (float*)d_out;

  // workspace layout:
  //   x   f32  [NNODES][256]   (residual carrier)        134.2 MB
  //   wt  bf16 transposed weights                          0.5 MB
  //   xb  bf16 [NNODES][256]   (GEMM A operand)           67.1 MB
  float* x = (float*)d_ws;
  unsigned short* wt_in  = (unsigned short*)((char*)d_ws + (size_t)NNODES * HIDDIM * 4);
  unsigned short* wt_blk = wt_in + INDIM * HIDDIM;                 // 3 x [256][256]
  unsigned short* wt_out = wt_blk + 3 * HIDDIM * HIDDIM;           // [128][256]
  unsigned short* xb     = wt_out + HIDDIM * OUTDIM;

  // weight conversion (bf16, transposed)
  wconv_kernel<<<(INDIM * HIDDIM + 255) / 256, 256, 0, stream>>>(W_in, wt_in, INDIM, HIDDIM);
  for (int k = 0; k < 3; ++k)
    wconv_kernel<<<(HIDDIM * HIDDIM + 255) / 256, 256, 0, stream>>>(
        Wb + (size_t)k * HIDDIM * HIDDIM, wt_blk + (size_t)k * HIDDIM * HIDDIM, HIDDIM, HIDDIM);
  wconv_kernel<<<(HIDDIM * OUTDIM + 255) / 256, 256, 0, stream>>>(W_out, wt_out, HIDDIM, OUTDIM);

  // stem: x = relu(LN(H @ W_in + b))  (A = f32 H, converted on the fly)
  mlp_layer_kernel<<<NNODES / 64, 128, 0, stream>>>(
      H, (const unsigned int*)0, INDIM, INDIM, (const unsigned int*)wt_in,
      b_in, g_in, be_in, x, xb, 0);
  // residual blocks: A = bf16 xb, residual from f32 x
  for (int k = 0; k < 3; ++k)
    mlp_layer_kernel<<<NNODES / 64, 128, 0, stream>>>(
        x, (const unsigned int*)xb, HIDDIM, HIDDIM,
        (const unsigned int*)(wt_blk + (size_t)k * HIDDIM * HIDDIM),
        bb + k * HIDDIM, gb + k * HIDDIM, betab + k * HIDDIM, x, xb, 1);
  // head + per-graph reduction (fused)
  head_reduce_kernel<<<NGRAPH, 128, 0, stream>>>(
      (const unsigned int*)xb, (const unsigned int*)wt_out, b_out, g_out, be_out, out);
}
